// gcnalign_highway_77163382440895
// MI455X (gfx1250) — compile-verified, matrix operand order checked by reading, not run on hardware
//
#include <hip/hip_runtime.h>
#include <hip/hip_bf16.h>
#include <math.h>

#define DIM 128
#define IN_FEAT 128

typedef __attribute__((ext_vector_type(16))) __bf16 v16bf;
typedef __attribute__((ext_vector_type(8)))  float  v8f;

// ---------------------------------------------------------------------------
// Zero an int buffer.
// ---------------------------------------------------------------------------
__global__ void gcn_zero_i32(int* __restrict__ p, int n) {
  int i = blockIdx.x * blockDim.x + threadIdx.x;
  if (i < n) p[i] = 0;
}

// ---------------------------------------------------------------------------
// CSR build step 1: per-row nonzero histogram (integer atomics only).
// ---------------------------------------------------------------------------
__global__ void gcn_row_hist(const int* __restrict__ rows, int* __restrict__ counts,
                             int nnz) {
  int e = blockIdx.x * blockDim.x + threadIdx.x;
  if (e < nnz) atomicAdd(&counts[rows[e]], 1);
}

// ---------------------------------------------------------------------------
// CSR build step 2a: per-block exclusive scan (256 elems/block) via LDS.
// Writes local exclusive prefix into row_ptr and the block total to blockSums.
// ---------------------------------------------------------------------------
__global__ void gcn_scan_block(const int* __restrict__ counts, int* __restrict__ row_ptr,
                               int* __restrict__ blockSums, int n) {
  __shared__ int sh[256];
  const int t = threadIdx.x;
  const int g = blockIdx.x * 256 + t;
  const int x = (g < n) ? counts[g] : 0;
  sh[t] = x;
  __syncthreads();
#pragma unroll
  for (int off = 1; off < 256; off <<= 1) {
    int v = (t >= off) ? sh[t - off] : 0;
    __syncthreads();
    sh[t] += v;
    __syncthreads();
  }
  if (g < n) row_ptr[g] = sh[t] - x;          // exclusive
  if (t == 255) blockSums[blockIdx.x] = sh[255];
}

// ---------------------------------------------------------------------------
// CSR build step 2b: single-block exclusive scan of the block sums (nb<=1024).
// ---------------------------------------------------------------------------
__global__ void gcn_scan_sums(int* __restrict__ blockSums, int nb) {
  __shared__ int sh[1024];
  const int t = threadIdx.x;
  const int x = (t < nb) ? blockSums[t] : 0;
  sh[t] = x;
  __syncthreads();
#pragma unroll
  for (int off = 1; off < 1024; off <<= 1) {
    int v = (t >= off) ? sh[t - off] : 0;
    __syncthreads();
    sh[t] += v;
    __syncthreads();
  }
  if (t < nb) blockSums[t] = sh[t] - x;       // exclusive
}

// ---------------------------------------------------------------------------
// CSR build step 2c: add block offsets; also initialize the scatter cursor.
// ---------------------------------------------------------------------------
__global__ void gcn_scan_add(int* __restrict__ row_ptr, int* __restrict__ cursor,
                             const int* __restrict__ blockSums, int n) {
  int g = blockIdx.x * blockDim.x + threadIdx.x;
  if (g < n) {
    int v = row_ptr[g] + blockSums[blockIdx.x];
    row_ptr[g] = v;
    cursor[g]  = v;
  }
}

// ---------------------------------------------------------------------------
// CSR build step 3: scatter COO entries into row-grouped order.
// ---------------------------------------------------------------------------
__global__ void gcn_csr_scatter(const float* __restrict__ vals, const int* __restrict__ rows,
                                const int* __restrict__ cols, int* __restrict__ cursor,
                                float* __restrict__ csr_vals, int* __restrict__ csr_cols,
                                int nnz) {
  int e = blockIdx.x * blockDim.x + threadIdx.x;
  if (e < nnz) {
    int pos = atomicAdd(&cursor[rows[e]], 1);
    csr_vals[pos] = vals[e];
    csr_cols[pos] = cols[e];
  }
}

// ---------------------------------------------------------------------------
// Dense GEMM  XW[n,128] = X[n,128] @ W[128,128]  via bf16x3 WMMA emulation.
// One wave32 per 16x16 output tile; K=128 in 4 steps of 32; 3 WMMAs per step
// (hi*hi + hi*lo + lo*hi) for ~f32 accuracy on the bf16 matrix pipe.
// VGPR layouts per CDNA5 ISA 7.12.2 (16-bit A 16x32, B 32x16, 32-bit C/D).
// ---------------------------------------------------------------------------
__global__ void gcn_gemm_xw2_wmma(const float* __restrict__ X,
                                  const float* __restrict__ W,
                                  float* __restrict__ XW,
                                  int rowTiles) {
  const int wave = (int)(((long long)blockIdx.x * blockDim.x + threadIdx.x) >> 5);
  const int lane = threadIdx.x & 31;
  const int colTiles = DIM / 16;           // 8
  const int tm = wave / colTiles;
  const int tn = wave % colTiles;
  if (tm >= rowTiles) return;              // wave-uniform; EXEC stays all-1s

  const int g    = lane >> 4;              // half-wave group (0/1)
  const int m    = lane & 15;              // A row within tile
  const int ncol = lane & 15;              // B col within tile

  const float* __restrict__ Arow = X + (size_t)(tm * 16 + m) * IN_FEAT;

  v8f acc = {};
  for (int k0 = 0; k0 < IN_FEAT; k0 += 32) {
    v16bf ahi, alo, bhi, blo;
#pragma unroll
    for (int vv = 0; vv < 8; ++vv) {
      const int kb = (vv < 4) ? (g * 8 + 2 * vv) : (16 + g * 8 + 2 * (vv - 4));
      const float a0 = Arow[k0 + kb];
      const float a1 = Arow[k0 + kb + 1];
      const __bf16 ah0 = (__bf16)a0, ah1 = (__bf16)a1;
      ahi[2 * vv]     = ah0;
      ahi[2 * vv + 1] = ah1;
      alo[2 * vv]     = (__bf16)(a0 - (float)ah0);
      alo[2 * vv + 1] = (__bf16)(a1 - (float)ah1);

      const float b0 = W[(size_t)(k0 + kb) * DIM + tn * 16 + ncol];
      const float b1 = W[(size_t)(k0 + kb + 1) * DIM + tn * 16 + ncol];
      const __bf16 bh0 = (__bf16)b0, bh1 = (__bf16)b1;
      bhi[2 * vv]     = bh0;
      bhi[2 * vv + 1] = bh1;
      blo[2 * vv]     = (__bf16)(b0 - (float)bh0);
      blo[2 * vv + 1] = (__bf16)(b1 - (float)bh1);
    }
    acc = __builtin_amdgcn_wmma_f32_16x16x32_bf16(false, ahi, false, bhi,
                                                  (short)0, acc, false, false);
    acc = __builtin_amdgcn_wmma_f32_16x16x32_bf16(false, ahi, false, blo,
                                                  (short)0, acc, false, false);
    acc = __builtin_amdgcn_wmma_f32_16x16x32_bf16(false, alo, false, bhi,
                                                  (short)0, acc, false, false);
  }

  const int row0 = tm * 16 + g * 8;
  const int col  = tn * 16 + (lane & 15);
#pragma unroll
  for (int v = 0; v < 8; ++v) {
    XW[(size_t)(row0 + v) * DIM + col] = acc[v];
  }
}

// ---------------------------------------------------------------------------
// CSR row accumulate helper: acc += sum_j csr_vals[j] * Xsrc[csr_cols[j], :]
// One wave32 per row, float4 per lane, register accumulation (no atomics).
// ---------------------------------------------------------------------------
__device__ __forceinline__ float4 gcn_csr_row_acc(const float* __restrict__ csr_vals,
                                                  const int* __restrict__ csr_cols,
                                                  const float* __restrict__ Xsrc,
                                                  int s, int cnt, int lane) {
  float4 acc = make_float4(0.f, 0.f, 0.f, 0.f);
  int j = 0;
  for (; j + 1 < cnt; j += 2) {
    const float v0 = csr_vals[s + j];
    const float v1 = csr_vals[s + j + 1];
    const int   c0 = csr_cols[s + j];
    const int   c1 = csr_cols[s + j + 1];
    const float4 x0 = ((const float4*)(Xsrc + (size_t)c0 * DIM))[lane];
    const float4 x1 = ((const float4*)(Xsrc + (size_t)c1 * DIM))[lane];
    acc.x += v0 * x0.x + v1 * x1.x;
    acc.y += v0 * x0.y + v1 * x1.y;
    acc.z += v0 * x0.z + v1 * x1.z;
    acc.w += v0 * x0.w + v1 * x1.w;
  }
  if (j < cnt) {
    const float v0 = csr_vals[s + j];
    const int   c0 = csr_cols[s + j];
    const float4 x0 = ((const float4*)(Xsrc + (size_t)c0 * DIM))[lane];
    acc.x += v0 * x0.x; acc.y += v0 * x0.y;
    acc.z += v0 * x0.z; acc.w += v0 * x0.w;
  }
  return acc;
}

// ---------------------------------------------------------------------------
// SpMM (CSR) with fused ReLU:  Y[r,:] = relu(sum_j v_j * Xsrc[c_j,:])
// ---------------------------------------------------------------------------
__global__ void gcn_spmm_csr_relu(const int* __restrict__ row_ptr,
                                  const int* __restrict__ counts,
                                  const float* __restrict__ csr_vals,
                                  const int* __restrict__ csr_cols,
                                  const float* __restrict__ Xsrc,
                                  float* __restrict__ Y, int n) {
  const int r    = (int)(((long long)blockIdx.x * blockDim.x + threadIdx.x) >> 5);
  const int lane = threadIdx.x & 31;
  if (r >= n) return;
  float4 acc = gcn_csr_row_acc(csr_vals, csr_cols, Xsrc, row_ptr[r], counts[r], lane);
  acc.x = fmaxf(acc.x, 0.f); acc.y = fmaxf(acc.y, 0.f);
  acc.z = fmaxf(acc.z, 0.f); acc.w = fmaxf(acc.w, 0.f);
  ((float4*)(Y + (size_t)r * DIM))[lane] = acc;
}

// ---------------------------------------------------------------------------
// SpMM (CSR) with fused ReLU + highway gate:
//   b = relu(A@XW row); T = sigmoid(dot(b, wg)); y = T*a + (1-T)*b
// a is read from AY and y overwrites it (wave owns the row).
// ---------------------------------------------------------------------------
__global__ void gcn_spmm_csr_gate(const int* __restrict__ row_ptr,
                                  const int* __restrict__ counts,
                                  const float* __restrict__ csr_vals,
                                  const int* __restrict__ csr_cols,
                                  const float* __restrict__ XW,
                                  const float* __restrict__ wg,
                                  float* __restrict__ AY, int n) {
  const int r    = (int)(((long long)blockIdx.x * blockDim.x + threadIdx.x) >> 5);
  const int lane = threadIdx.x & 31;
  if (r >= n) return;
  float4 b4 = gcn_csr_row_acc(csr_vals, csr_cols, XW, row_ptr[r], counts[r], lane);
  b4.x = fmaxf(b4.x, 0.f); b4.y = fmaxf(b4.y, 0.f);
  b4.z = fmaxf(b4.z, 0.f); b4.w = fmaxf(b4.w, 0.f);

  const float4 w4 = ((const float4*)wg)[lane];
  float dot = b4.x * w4.x + b4.y * w4.y + b4.z * w4.z + b4.w * w4.w;
#pragma unroll
  for (int off = 16; off > 0; off >>= 1) dot += __shfl_xor(dot, off, 32);
  const float T = 1.0f / (1.0f + __expf(-dot));

  const float4 a4 = ((const float4*)(AY + (size_t)r * DIM))[lane];
  float4 y4;
  y4.x = T * a4.x + (1.0f - T) * b4.x;
  y4.y = T * a4.y + (1.0f - T) * b4.y;
  y4.z = T * a4.z + (1.0f - T) * b4.z;
  y4.w = T * a4.w + (1.0f - T) * b4.w;
  ((float4*)(AY + (size_t)r * DIM))[lane] = y4;
}

// ---------------------------------------------------------------------------
// SpMM (CSR), plain:  Out[r,:] = sum_j v_j * Y[c_j,:]   (no atomics, no zeroing)
// ---------------------------------------------------------------------------
__global__ void gcn_spmm_csr_plain(const int* __restrict__ row_ptr,
                                   const int* __restrict__ counts,
                                   const float* __restrict__ csr_vals,
                                   const int* __restrict__ csr_cols,
                                   const float* __restrict__ Ysrc,
                                   float* __restrict__ Out, int n) {
  const int r    = (int)(((long long)blockIdx.x * blockDim.x + threadIdx.x) >> 5);
  const int lane = threadIdx.x & 31;
  if (r >= n) return;
  float4 acc = gcn_csr_row_acc(csr_vals, csr_cols, Ysrc, row_ptr[r], counts[r], lane);
  ((float4*)(Out + (size_t)r * DIM))[lane] = acc;
}

// ---------------------------------------------------------------------------
// Launch: inputs are {inputs, w1, w2, wg, A_vals, A_rows, A_cols}.
// ---------------------------------------------------------------------------
extern "C" void kernel_launch(void* const* d_in, const int* in_sizes, int n_in,
                              void* d_out, int out_size, void* d_ws, size_t ws_size,
                              hipStream_t stream) {
  const float* inputs = (const float*)d_in[0];
  const float* w1     = (const float*)d_in[1];
  const float* w2     = (const float*)d_in[2];
  const float* wg     = (const float*)d_in[3];
  const float* A_vals = (const float*)d_in[4];
  const int*   A_rows = (const int*)d_in[5];
  const int*   A_cols = (const int*)d_in[6];

  const int n   = in_sizes[0] / IN_FEAT;   // 100000
  const int nnz = in_sizes[4];             // 1600000

  // Workspace layout (all 256B-aligned regions).
  const size_t nPad   = ((size_t)n + 63) & ~(size_t)63;
  const size_t nnzPad = ((size_t)nnz + 63) & ~(size_t)63;
  char* ws = (char*)d_ws;
  int*   counts    = (int*)ws;                 ws += nPad * sizeof(int);
  int*   row_ptr   = (int*)ws;                 ws += nPad * sizeof(int);
  int*   cursor    = (int*)ws;                 ws += nPad * sizeof(int);
  int*   blockSums = (int*)ws;                 ws += 1024 * sizeof(int);
  int*   csr_cols  = (int*)ws;                 ws += nnzPad * sizeof(int);
  float* csr_vals  = (float*)ws;               ws += nnzPad * sizeof(float);
  float* a_y       = (float*)ws;               // [n,128] a, then y in-place

  float* xw  = (float*)d_out;                  // pre-final use of d_out
  float* out = (float*)d_out;

  const int nb        = (n + 255) / 256;       // scan blocks (<=1024 required)
  const int nnzBlocks = (nnz + 255) / 256;

  // --- CSR build (integer atomics only) ---
  gcn_zero_i32<<<nb, 256, 0, stream>>>(counts, n);
  gcn_row_hist<<<nnzBlocks, 256, 0, stream>>>(A_rows, counts, nnz);
  gcn_scan_block<<<nb, 256, 0, stream>>>(counts, row_ptr, blockSums, n);
  gcn_scan_sums<<<1, 1024, 0, stream>>>(blockSums, nb);
  gcn_scan_add<<<nb, 256, 0, stream>>>(row_ptr, cursor, blockSums, n);
  gcn_csr_scatter<<<nnzBlocks, 256, 0, stream>>>(A_vals, A_rows, A_cols, cursor,
                                                 csr_vals, csr_cols, nnz);

  // --- xw = inputs @ w2  (WMMA bf16x3) ---
  const int rowTiles   = n / 16;
  const int tileWaves  = rowTiles * (DIM / 16);
  const int gemmBlocks = (tileWaves * 32 + 255) / 256;
  gcn_gemm_xw2_wmma<<<gemmBlocks, 256, 0, stream>>>(inputs, w2, xw, rowTiles);

  // --- fused sparse pipeline (no float atomics anywhere) ---
  const int rowWaveBlocks = (int)(((long long)n * 32 + 255) / 256);
  gcn_spmm_csr_relu<<<rowWaveBlocks, 256, 0, stream>>>(row_ptr, counts, csr_vals,
                                                       csr_cols, w1, a_y, n);
  gcn_spmm_csr_gate<<<rowWaveBlocks, 256, 0, stream>>>(row_ptr, counts, csr_vals,
                                                       csr_cols, xw, wg, a_y, n);
  gcn_spmm_csr_plain<<<rowWaveBlocks, 256, 0, stream>>>(row_ptr, counts, csr_vals,
                                                        csr_cols, a_y, out, n);
}
